// VectorQuantizer_44169443672822
// MI455X (gfx1250) — matrix operand order repaired and checked by reading, not compiled, not run on hardware
//
#include <hip/hip_runtime.h>
#include <hip/hip_bf16.h>
#include <math.h>

// ---------------- problem constants ----------------
#define N_TOK   131072
#define N_EMB   1024
#define EMB_D   64
#define COMMIT  0.25f

// LDS codebook row stride in bf16 units: 72 halfwords = 144 bytes = 36 dwords.
// 36*r mod 64 is distinct for r=0..15 -> conflict-free ds_load_b128 across lanes.
#define CB_STRIDE 72
// Distance bias making d' = (||c||^2 + BIAS) - 2 z.c strictly positive so IEEE
// f32 bit patterns order like unsigned ints (|2 z.c| <= ~0.2 for this data).
#define DIST_BIAS 0.5f

typedef __attribute__((ext_vector_type(16))) __bf16 v16bf;
typedef __attribute__((ext_vector_type(8)))  float  v8f;

union BFrag { uint4 q[2]; v16bf v; };

__device__ __forceinline__ unsigned pk2(float a, float b) {
    union { __bf16 h[2]; unsigned u; } v;
    v.h[0] = (__bf16)a; v.h[1] = (__bf16)b;
    return v.u;
}
__device__ __forceinline__ unsigned umin2(unsigned a, unsigned b) { return a < b ? a : b; }

// ---------------- kernel 0: zero workspace (counts[1024] + sse) ----------------
__global__ void vq_zero_kernel(int* __restrict__ counts, float* __restrict__ sse) {
    int t = blockIdx.x * blockDim.x + threadIdx.x;
    if (t < N_EMB) counts[t] = 0;
    if (t == N_EMB) *sse = 0.0f;
}

// ---------------- kernel 1: WMMA distance + argmin ----------------
// grid = N_TOK/256 blocks of 256 threads (8 waves, 32 tokens per wave).
// dynamic LDS: bf16 codebook [1024][72] + f32 biased norms [1024]
__launch_bounds__(256, 1)
__global__ void vq_argmin_kernel(const float* __restrict__ z_e,
                                 const float* __restrict__ cb,
                                 float* __restrict__ out_idx_f) {
    extern __shared__ __align__(16) char smem[];
    unsigned short* lds_cb   = (unsigned short*)smem;                       // bf16 bits
    float*          lds_norm = (float*)(smem + N_EMB * CB_STRIDE * 2);

    const int tid = threadIdx.x;

    // ---- stage codebook -> LDS (f32 -> bf16) and biased ||c||^2 in f32 ----
    for (int c = tid; c < N_EMB; c += 256) {
        const float4* src = (const float4*)(cb + c * EMB_D);
        uint4* dst = (uint4*)(lds_cb + c * CB_STRIDE);
        float nrm = 0.0f;
        #pragma unroll
        for (int q = 0; q < 8; ++q) {
            float4 f0 = src[2 * q + 0];
            float4 f1 = src[2 * q + 1];
            nrm += f0.x * f0.x + f0.y * f0.y + f0.z * f0.z + f0.w * f0.w;
            nrm += f1.x * f1.x + f1.y * f1.y + f1.z * f1.z + f1.w * f1.w;
            uint4 u;
            u.x = pk2(f0.x, f0.y); u.y = pk2(f0.z, f0.w);
            u.z = pk2(f1.x, f1.y); u.w = pk2(f1.z, f1.w);
            dst[q] = u;                       // ds_store_b128, 16B aligned
        }
        lds_norm[c] = nrm + DIST_BIAS;
    }
    __syncthreads();

    const int wave = tid >> 5;
    const int lane = tid & 31;
    const int l15  = lane & 15;
    const int hf   = lane >> 4;               // which half of the wave
    const int tok0 = blockIdx.x * 256 + wave * 32;

    // ---- build two A tiles (32 tokens x 64 dims, bf16), ISA 16-bit A layout ----
    // lane<16:  V0-3 = K 0..7,  V4-7 = K 16..23   (per 32-K chunk)
    // lane>=16: V0-3 = K 8..15, V4-7 = K 24..31
    v16bf a0, a1, a2, a3;
    {
        const float* zr0 = z_e + (size_t)(tok0 + l15) * EMB_D + hf * 8;
        const float* zr1 = z_e + (size_t)(tok0 + 16 + l15) * EMB_D + hf * 8;
        #pragma unroll
        for (int i = 0; i < 8; ++i) {
            a0[i]     = (__bf16)zr0[i];        a0[8 + i] = (__bf16)zr0[16 + i];
            a1[i]     = (__bf16)zr0[32 + i];   a1[8 + i] = (__bf16)zr0[48 + i];
            a2[i]     = (__bf16)zr1[i];        a2[8 + i] = (__bf16)zr1[16 + i];
            a3[i]     = (__bf16)zr1[32 + i];   a3[8 + i] = (__bf16)zr1[48 + i];
        }
    }

    // Packed argmin keys: (f32 bits of biased dist & ~1023) | code.
    unsigned key0[8], key1[8];
    #pragma unroll
    for (int i = 0; i < 8; ++i) { key0[i] = 0xFFFFFFFFu; key1[i] = 0xFFFFFFFFu; }

    // ---- software-pipelined sweep over 64 code tiles of 16 codes each ----
    // B layout (32x16 bf16, K x N): lane<16 col=l15 K 0..7,16..23; lane>=16 K 8..15,24..31.
    BFrag b0c, b1c, b0n, b1n;
    float cn_c, cn_n;
    {
        const uint4* r = (const uint4*)(lds_cb + l15 * CB_STRIDE + hf * 8);
        b0c.q[0] = r[0]; b0c.q[1] = r[2];      // dims hf*8 .. +8, 16+hf*8 .. +8
        const uint4* r1 = (const uint4*)(lds_cb + l15 * CB_STRIDE + 32 + hf * 8);
        b1c.q[0] = r1[0]; b1c.q[1] = r1[2];    // dims 32+hf*8 .., 48+hf*8 ..
        cn_c = lds_norm[l15];
    }

    #pragma unroll 2
    for (int kt = 0; kt < N_EMB / 16; ++kt) {
        // prefetch next tile's B fragments while this tile computes
        if (kt + 1 < N_EMB / 16) {
            const int nc = (kt + 1) * 16 + l15;
            const uint4* r = (const uint4*)(lds_cb + nc * CB_STRIDE + hf * 8);
            b0n.q[0] = r[0]; b0n.q[1] = r[2];
            const uint4* r1 = (const uint4*)(lds_cb + nc * CB_STRIDE + 32 + hf * 8);
            b1n.q[0] = r1[0]; b1n.q[1] = r1[2];
            cn_n = lds_norm[nc];
        }

        v8f acc0 = {0.f, 0.f, 0.f, 0.f, 0.f, 0.f, 0.f, 0.f};
        v8f acc1 = {0.f, 0.f, 0.f, 0.f, 0.f, 0.f, 0.f, 0.f};
        acc0 = __builtin_amdgcn_wmma_f32_16x16x32_bf16(false, a0, false, b0c.v,
                                                       (short)0, acc0, false, false);
        acc0 = __builtin_amdgcn_wmma_f32_16x16x32_bf16(false, a1, false, b1c.v,
                                                       (short)0, acc0, false, false);
        acc1 = __builtin_amdgcn_wmma_f32_16x16x32_bf16(false, a2, false, b0c.v,
                                                       (short)0, acc1, false, false);
        acc1 = __builtin_amdgcn_wmma_f32_16x16x32_bf16(false, a3, false, b1c.v,
                                                       (short)0, acc1, false, false);

        const unsigned code = (unsigned)(kt * 16 + l15);
        #pragma unroll
        for (int i = 0; i < 8; ++i) {
            float d0 = fmaf(-2.0f, acc0[i], cn_c);           // biased, > 0
            float d1 = fmaf(-2.0f, acc1[i], cn_c);
            key0[i] = umin2(key0[i], (__float_as_uint(d0) & 0xFFFFFC00u) | code);
            key1[i] = umin2(key1[i], (__float_as_uint(d1) & 0xFFFFFC00u) | code);
        }

        b0c = b0n; b1c = b1n; cn_c = cn_n;
    }

    // ---- cross-lane key-min within each 16-lane half (codes live across lanes) ----
    #pragma unroll
    for (int off = 1; off < 16; off <<= 1) {
        #pragma unroll
        for (int i = 0; i < 8; ++i) {
            key0[i] = umin2(key0[i], (unsigned)__shfl_xor((int)key0[i], off, 32));
            key1[i] = umin2(key1[i], (unsigned)__shfl_xor((int)key1[i], off, 32));
        }
    }

    // D layout: VGPR i holds row M=i (lanes 0-15) / M=8+i (lanes 16-31)
    if (l15 == 0) {
        #pragma unroll
        for (int i = 0; i < 8; ++i) {
            out_idx_f[tok0 + hf * 8 + i]      = (float)(key0[i] & 1023u);
            out_idx_f[tok0 + 16 + hf * 8 + i] = (float)(key1[i] & 1023u);
        }
    }
}

// ---------------- kernel 2: gather z_q, SSE, histogram ----------------
__launch_bounds__(256)
__global__ void vq_gather_kernel(const float* __restrict__ z_e,
                                 const float* __restrict__ cb,
                                 const float* __restrict__ idx_f,
                                 float* __restrict__ z_q,
                                 int* __restrict__ counts,
                                 float* __restrict__ sse) {
    __shared__ float red[256];
    const int t = blockIdx.x * 256 + threadIdx.x;
    const int idx = (int)idx_f[t];

    const float4* c = (const float4*)(cb + (size_t)idx * EMB_D);
    const float4* z = (const float4*)(z_e + (size_t)t * EMB_D);
    float4*       o = (float4*)(z_q + (size_t)t * EMB_D);

    float s = 0.0f;
    #pragma unroll
    for (int q = 0; q < 16; ++q) {
        float4 cv = c[q];
        float4 zv = z[q];
        o[q] = cv;                             // z_q_st forward value == z_q
        float dx = cv.x - zv.x, dy = cv.y - zv.y;
        float dz = cv.z - zv.z, dw = cv.w - zv.w;
        s += dx * dx + dy * dy + dz * dz + dw * dw;
    }
    atomicAdd(&counts[idx], 1);

    red[threadIdx.x] = s;
    __syncthreads();
    for (int off = 128; off > 0; off >>= 1) {
        if (threadIdx.x < off) red[threadIdx.x] += red[threadIdx.x + off];
        __syncthreads();
    }
    if (threadIdx.x == 0) atomicAdd(sse, red[0]);
}

// ---------------- kernel 3: loss + perplexity ----------------
__launch_bounds__(1024)
__global__ void vq_finalize_kernel(const int* __restrict__ counts,
                                   const float* __restrict__ sse,
                                   float* __restrict__ scalars) {
    __shared__ float red[1024];
    const int t = threadIdx.x;
    float p = (float)counts[t] * (1.0f / (float)N_TOK);
    red[t] = p * logf(p + 1e-10f);
    __syncthreads();
    for (int off = 512; off > 0; off >>= 1) {
        if (t < off) red[t] += red[t + off];
        __syncthreads();
    }
    if (t == 0) {
        float mse = sse[0] * (1.0f / ((float)N_TOK * (float)EMB_D));
        scalars[0] = (1.0f + COMMIT) * mse;    // codebook + commit losses (equal fwd)
        scalars[1] = expf(-red[0]);
    }
}

// ---------------- launcher ----------------
extern "C" void kernel_launch(void* const* d_in, const int* in_sizes, int n_in,
                              void* d_out, int out_size, void* d_ws, size_t ws_size,
                              hipStream_t stream) {
    (void)in_sizes; (void)n_in; (void)out_size; (void)ws_size;
    const float* z_e = (const float*)d_in[0];
    const float* cb  = (const float*)d_in[1];

    float* out     = (float*)d_out;
    float* z_q     = out;                                  // [N_TOK * EMB_D]
    float* scalars = out + (size_t)N_TOK * EMB_D;          // [loss, perplexity]
    float* idx_f   = scalars + 2;                          // [N_TOK] indices (float)

    int*   counts = (int*)d_ws;                            // 1024 ints
    float* sse    = (float*)((char*)d_ws + 4096);          // 1 float

    vq_zero_kernel<<<5, 256, 0, stream>>>(counts, sse);

    const size_t lds_bytes = (size_t)N_EMB * CB_STRIDE * 2 + (size_t)N_EMB * 4; // 148 KB
    vq_argmin_kernel<<<N_TOK / 256, 256, lds_bytes, stream>>>(z_e, cb, idx_f);

    vq_gather_kernel<<<N_TOK / 256, 256, 0, stream>>>(z_e, cb, idx_f, z_q, counts, sse);

    vq_finalize_kernel<<<1, 1024, 0, stream>>>(counts, sse, scalars);
}
